// UpperLevelIMTSP_7670811590990
// MI455X (gfx1250) — matrix operand order repaired and checked by reading, not compiled, not run on hardware
//
#include <hip/hip_runtime.h>
#include <hip/hip_bf16.h>
#include <math.h>

// ---------------------------------------------------------------------------
// IMTSP upper-level forward for MI455X (gfx1250, wave32, WMMA).
// - All matmuls via v_wmma_f32_16x16x32_f16 (f16 operands, f32 accumulate).
// - Weights pre-packed into WMMA lane order; each block async-stages its
//   B panel into LDS with global_load_async_to_lds_b128 + s_wait_asynccnt.
// - V is written transposed so the P@V B-fragment is two contiguous b128s.
// Workload is HBM-bound (~1.5-2 GB streamed @ 23.3 TB/s).
// ---------------------------------------------------------------------------

typedef __attribute__((ext_vector_type(16))) _Float16 v16h;
typedef __attribute__((ext_vector_type(8)))  _Float16 h8;
typedef __attribute__((ext_vector_type(8)))  float    v8f;

constexpr int   Bg   = 256;    // batch (graphs)
constexpr int   Nn   = 1024;   // nodes per graph
constexpr int   NDd  = 16;     // depots / agents
constexpr int   Dd   = 64;     // graph embed
constexpr int   SHh  = 256;    // surrogate hidden
constexpr float CLIPf = 10.0f;
constexpr float BN_EPSf = 1e-5f;
constexpr long  Mbig = (long)Bg * Nn;     // 262144 rows

// ---------------------------- tiny utility kernels -------------------------

__global__ void k_zero_f32(float* p, long n) {
  long i = (long)blockIdx.x * blockDim.x + threadIdx.x;
  if (i < n) p[i] = 0.0f;
}
__global__ void k_zero_f16(_Float16* p, long n) {
  long i = (long)blockIdx.x * blockDim.x + threadIdx.x;
  if (i < n) p[i] = (_Float16)0.0f;
}
__global__ void k_copy_f32(const float* s, float* d, long n) {
  long i = (long)blockIdx.x * blockDim.x + threadIdx.x;
  if (i < n) d[i] = s[i];
}
// f32 -> f16 with scale (scale=1 plain convert, 1/N for the mean-pool)
__global__ void k_cvt_f16(const float* s, _Float16* d, long n, float scale) {
  long i = (long)blockIdx.x * blockDim.x + threadIdx.x;
  if (i < n) d[i] = (_Float16)(s[i] * scale);
}
// zero-pad extra columns: (rows x sc) f32 -> (rows x dc) f16
__global__ void k_padcols(const float* s, _Float16* d, long rows, int sc, int dc) {
  long i = (long)blockIdx.x * blockDim.x + threadIdx.x;
  if (i >= rows * dc) return;
  long r = i / dc; int c = (int)(i % dc);
  d[i] = (c < sc) ? (_Float16)s[r * sc + c] : (_Float16)0.0f;
}

// ------------------- weight repack into WMMA fragment order ----------------
// src: (Ksrc x Nc) f32 row-major.  dst: fragments f = nt*(Kp/32) + ktidx,
// each fragment = 32 lanes x 16 halves (lane-contiguous 32B runs).
// Lane l of fragment (ktidx, nt) holds W[ktidx*32 + (l>>4)*16 + i][nt*16 + (l&15)]
// for i = 0..15 — contiguous k order, matching the A-fragment loads.  K rows beyond
// Ksrc are zero (K padding for AD=16 and the 1008-wide surrogate input).
__global__ void k_packW(const float* src, _Float16* dst, int Ksrc, int Kp, int Nc) {
  long i = (long)blockIdx.x * blockDim.x + threadIdx.x;
  if (i >= (long)Kp * Nc) return;
  int e    = (int)(i & 15);
  int lane = (int)((i >> 4) & 31);
  long f   = i >> 9;
  int kfrags = Kp >> 5;
  int ktidx = (int)(f % kfrags);
  int nt    = (int)(f / kfrags);
  int k = ktidx * 32 + (lane >> 4) * 16 + e;
  int c = nt * 16 + (lane & 15);
  dst[i] = (k < Ksrc) ? (_Float16)src[(long)k * Nc + c] : (_Float16)0.0f;
}

// ---------------------------- generic WMMA GEMM ----------------------------
// C(M x Nout) = act( A(M x K, f16) * W(K x Nout, packed f16) + bias )
// grid.x = M/64 (4 waves stacked in M), grid.y = Nout/16, block = 128.
// act: 0 none, 1 relu, 2 tanh.  tposeV: write f16 out as Vt[b][col][n].
__global__ __launch_bounds__(128) void k_gemm(
    const _Float16* __restrict__ A, int lda,
    const _Float16* __restrict__ Wp,
    const float* __restrict__ bias,
    float* __restrict__ Cf, _Float16* __restrict__ Ch, int ldc,
    int K, int act, int tposeV) {
  __shared__ _Float16 wsm[16384];          // up to K=1024 of B fragments (32KB)
  const int lane = threadIdx.x & 31;
  const int wave = threadIdx.x >> 5;
  const int r    = lane & 15;
  const int hi   = lane >> 4;
  const long mrow0 = ((long)blockIdx.x * 4 + wave) * 16;
  const int  n0    = blockIdx.y * 16;
  const long arow  = mrow0 + r;
  const int  col   = n0 + r;

  // Async-stage this block's contiguous B panel (shared by all 4 waves).
  const int  kfrags    = K >> 5;
  const long tilebytes = (long)kfrags * 1024;
  const char* gsrc = (const char*)Wp + (long)blockIdx.y * tilebytes;
  unsigned sbase = (unsigned)(size_t)(&wsm[0]);
  for (long ofs = (long)threadIdx.x * 16; ofs < tilebytes; ofs += 128 * 16) {
    unsigned lo = sbase + (unsigned)ofs;
    const void* gp = gsrc + ofs;
    asm volatile("global_load_async_to_lds_b128 %0, %1, off"
                 :: "v"(lo), "v"(gp) : "memory");
  }
  asm volatile("s_wait_asynccnt 0x0" ::: "memory");
  __syncthreads();

  const char* wtile = (const char*)wsm + (long)lane * 32;
  v8f acc = {};
  for (int kt = 0; kt < K; kt += 32) {
    // A fragment: lane r, halves [0..15]=K kb..kb+15,
    // kb = kt + hi*16 — two contiguous 16B loads.
    const _Float16* ap = A + arow * lda + kt + hi * 16;
    h8 alo = *(const h8*)(ap);
    h8 ahi = *(const h8*)(ap + 8);
    v16h af;
#pragma unroll
    for (int i = 0; i < 8; ++i) { af[i] = alo[i]; af[8 + i] = ahi[i]; }
    // B fragment: lane-contiguous 32B from LDS (ds_load_b128 x2).
    const h8* bp = (const h8*)(wtile + (long)(kt >> 5) * 1024);
    h8 blo = bp[0], bhi = bp[1];
    v16h bf;
#pragma unroll
    for (int i = 0; i < 8; ++i) { bf[i] = blo[i]; bf[8 + i] = bhi[i]; }
    acc = __builtin_amdgcn_wmma_f32_16x16x32_f16(false, af, false, bf,
                                                 (short)0, acc, false, false);
  }
  const float bcol = bias ? bias[col] : 0.0f;
#pragma unroll
  for (int j = 0; j < 8; ++j) {
    long row = mrow0 + hi * 8 + j;           // C layout: VGPR j -> M = hi*8+j
    float v = acc[j] + bcol;
    if (act == 1) v = fmaxf(v, 0.0f);
    else if (act == 2) v = tanhf(v);
    if (Cf) Cf[row * ldc + col] = v;
    if (Ch) {
      if (tposeV)  // V GEMM: store (b, d, n) so P@V B-fragments are contiguous
        Ch[((long)(row >> 10) * Dd + col) * Nn + (row & 1023)] = (_Float16)v;
      else
        Ch[row * ldc + col] = (_Float16)v;
    }
  }
}

// ---------------------- per-graph attention score GEMM ---------------------
// U[b, m, n] = f( scale * Q[b,m,:] . K[b,n,:] ).  Kᵀ[k][n] = K[n][k] is
// contiguous in k -> vector loads.  grid = (B, N/64), block = 128.
__global__ __launch_bounds__(128) void k_attn_scores(
    const _Float16* __restrict__ Q,   // (B*16, Kd)
    const _Float16* __restrict__ Km,  // (B*N , Kd)
    float* __restrict__ U,            // (B*16, N)
    int Kd, float scale, int do_tanh) {
  const int b    = blockIdx.x;
  const int lane = threadIdx.x & 31;
  const int wave = threadIdx.x >> 5;
  const int r = lane & 15, hi = lane >> 4;
  const int ntile = blockIdx.y * 4 + wave;
  const int node  = ntile * 16 + r;
  const _Float16* Qb = Q + (long)b * 16 * Kd;
  const _Float16* Kb = Km + (long)b * Nn * Kd;

  v8f acc = {};
  for (int kt = 0; kt < Kd; kt += 32) {
    const _Float16* ap = Qb + (long)r * Kd + kt + hi * 16;
    h8 alo = *(const h8*)(ap);
    h8 ahi = *(const h8*)(ap + 8);
    v16h af;
#pragma unroll
    for (int i = 0; i < 8; ++i) { af[i] = alo[i]; af[8 + i] = ahi[i]; }
    const _Float16* kp = Kb + (long)node * Kd + kt + hi * 16;
    h8 blo = *(const h8*)(kp);
    h8 bhi = *(const h8*)(kp + 8);
    v16h bf;
#pragma unroll
    for (int i = 0; i < 8; ++i) { bf[i] = blo[i]; bf[8 + i] = bhi[i]; }
    acc = __builtin_amdgcn_wmma_f32_16x16x32_f16(false, af, false, bf,
                                                 (short)0, acc, false, false);
  }
#pragma unroll
  for (int j = 0; j < 8; ++j) {
    int m = hi * 8 + j;
    float v = acc[j] * scale;
    if (do_tanh) v = tanhf(v) * CLIPf;
    U[((long)b * 16 + m) * Nn + ntile * 16 + r] = v;
  }
}

// --------------------------- per-graph P @ V GEMM --------------------------
// Out[b,m,d] = sum_n P[b,m,n] * Vt[b,d,n].  grid = B, block = 128 (4 dtiles).
__global__ __launch_bounds__(128) void k_attn_pv(
    const _Float16* __restrict__ P,   // (B*16, N)
    const _Float16* __restrict__ Vt,  // (B, D, N) transposed
    _Float16* __restrict__ Out) {     // (B*16, D)
  const int b    = blockIdx.x;
  const int lane = threadIdx.x & 31;
  const int ntile = threadIdx.x >> 5;   // 0..3
  const int r = lane & 15, hi = lane >> 4;
  const int col = ntile * 16 + r;       // output channel d
  const _Float16* Pb  = P  + (long)b * 16 * Nn;
  const _Float16* Vtb = Vt + (long)b * Dd * Nn;

  v8f acc = {};
  for (int kt = 0; kt < Nn; kt += 32) {
    const _Float16* ap = Pb + (long)r * Nn + kt + hi * 16;
    h8 alo = *(const h8*)(ap);
    h8 ahi = *(const h8*)(ap + 8);
    v16h af;
#pragma unroll
    for (int i = 0; i < 8; ++i) { af[i] = alo[i]; af[8 + i] = ahi[i]; }
    const _Float16* kp = Vtb + (long)col * Nn + kt + hi * 16;  // contiguous in n
    h8 blo = *(const h8*)(kp);
    h8 bhi = *(const h8*)(kp + 8);
    v16h bf;
#pragma unroll
    for (int i = 0; i < 8; ++i) { bf[i] = blo[i]; bf[8 + i] = bhi[i]; }
    acc = __builtin_amdgcn_wmma_f32_16x16x32_f16(false, af, false, bf,
                                                 (short)0, acc, false, false);
  }
#pragma unroll
  for (int j = 0; j < 8; ++j)
    Out[((long)b * 16 + hi * 8 + j) * Dd + col] = (_Float16)acc[j];
}

// ------------------------------ model-specific -----------------------------

__global__ void k_flag(const int* depots, float* flag) {
  flag[(long)blockIdx.x * Nn + depots[blockIdx.x * NDd + threadIdx.x]] = 1.0f;
}

// first linear: [x, y, depot_flag] @ W(3x64) + b
__global__ void k_lin1(const float* coords, const float* flag,
                       const float* W, const float* bias,
                       float* out, long total) {
  long i = (long)blockIdx.x * blockDim.x + threadIdx.x;
  if (i >= total) return;
  long rrow = i >> 6; int c = (int)(i & 63);
  float x = coords[rrow * 2], y = coords[rrow * 2 + 1], f = flag[rrow];
  out[i] = x * W[c] + y * W[64 + c] + f * W[128 + c] + bias[c];
}

// column sums / sum-of-squares over M rows of a (M x 64) matrix
__global__ __launch_bounds__(64) void k_colstats(const float* X, long M, float* stats) {
  int c = threadIdx.x;
  long r0 = (long)blockIdx.x * 1024;
  long r1 = r0 + 1024; if (r1 > M) r1 = M;
  float s = 0.0f, s2 = 0.0f;
  for (long r = r0; r < r1; ++r) { float v = X[r * 64 + c]; s += v; s2 += v * v; }
  atomicAdd(&stats[c], s);
  atomicAdd(&stats[64 + c], s2);
}
__global__ __launch_bounds__(64) void k_bnfinal(float* stats, float invM) {
  int c = threadIdx.x;
  float mu = stats[c] * invM;
  float var = stats[64 + c] * invM - mu * mu;
  stats[128 + c] = mu;
  stats[192 + c] = rsqrtf(var + BN_EPSf);
}
__global__ void k_bnapply(const float* X, const float* stats,
                          const float* g, const float* beta, long total,
                          int relu, float* of, _Float16* oh, float* addto) {
  long i = (long)blockIdx.x * blockDim.x + threadIdx.x;
  if (i >= total) return;
  int c = (int)(i & 63);
  float v = (X[i] - stats[128 + c]) * stats[192 + c] * g[c] + beta[c];
  if (relu) v = fmaxf(v, 0.0f);
  if (of) of[i] = v;
  if (oh) oh[i] = (_Float16)v;
  if (addto) addto[i] += v;
}

// per-graph per-channel sum over the N nodes
__global__ __launch_bounds__(64) void k_gsum(const float* X, float* out) {
  int b = blockIdx.x, d = threadIdx.x;
  const float* p = X + (long)b * Nn * 64 + d;
  float s = 0.0f;
  for (int n = 0; n < Nn; ++n) s += p[(long)n * 64];
  out[b * 64 + d] = s;
}

// GIN message: m = h + (sum - h)/(N-1), emitted in f16 for the next GEMM
__global__ void k_m16(const float* H, const float* gsum, _Float16* out, long total) {
  long i = (long)blockIdx.x * blockDim.x + threadIdx.x;
  if (i >= total) return;
  int d = (int)(i & 63);
  long row = i >> 6; int b = (int)(row >> 10);
  float h = H[i];
  out[i] = (_Float16)(h + (gsum[b * 64 + d] - h) * (1.0f / (Nn - 1)));
}

// fc = concat(ctx_bn, ve[depot]) in f16 — rows (b, j), 128 cols
__global__ __launch_bounds__(128) void k_fc16(const float* ctxbn, const float* v,
                                              const int* depots, _Float16* fc) {
  int b = blockIdx.x >> 4, j = blockIdx.x & 15, c = threadIdx.x;
  float val;
  if (c < 64) val = ctxbn[b * 64 + c];
  else {
    int node = depots[b * NDd + j];
    val = v[((long)b * Nn + node) * 64 + (c - 64)];
  }
  fc[((long)(b * 16 + j)) * 128 + c] = (_Float16)val;
}

// depot mask: column -inf, then diagonal 0 (depots are distinct per graph)
__global__ void k_mask(float* logits, const int* depots) {
  int b = blockIdx.x, j = threadIdx.x;
  int n = depots[b * NDd + j];
  for (int a = 0; a < 16; ++a)
    logits[((long)b * 16 + a) * Nn + n] = -__builtin_inff();
  __syncthreads();
  logits[((long)b * 16 + j) * Nn + n] = 0.0f;
}

// row softmax of U (B*16 rows x N), output f16 probabilities
__global__ __launch_bounds__(256) void k_softmax(const float* U, _Float16* P) {
  __shared__ float red[256];
  long row = blockIdx.x; int t = threadIdx.x;
  const float* u = U + row * Nn;
  float m = -__builtin_inff();
  for (int i = t; i < Nn; i += 256) m = fmaxf(m, u[i]);
  red[t] = m; __syncthreads();
  for (int s = 128; s > 0; s >>= 1) { if (t < s) red[t] = fmaxf(red[t], red[t + s]); __syncthreads(); }
  m = red[0]; __syncthreads();
  float s = 0.0f;
  for (int i = t; i < Nn; i += 256) s += expf(u[i] - m);
  red[t] = s; __syncthreads();
  for (int st = 128; st > 0; st >>= 1) { if (t < st) red[t] += red[t + st]; __syncthreads(); }
  float inv = 1.0f / red[0];
  for (int i = t; i < Nn; i += 256) P[row * Nn + i] = (_Float16)(expf(u[i] - m) * inv);
}

// Gumbel-argmax sampling + selected log-softmax over the 16 agents
__global__ __launch_bounds__(256) void k_sample(const float* logits,
                                                float* agent_out, float* logp_out) {
  long i = (long)blockIdx.x * 256 + threadIdx.x;
  if (i >= Mbig) return;
  int b = (int)(i >> 10), n = (int)(i & 1023);
  float l[16];
  float best = -__builtin_inff(); int bi = 0;
  float m = -__builtin_inff();
  for (int a = 0; a < 16; ++a) {
    float v = logits[((long)b * 16 + a) * Nn + n];
    l[a] = v; m = fmaxf(m, v);
    unsigned x = (unsigned)(i * 16 + a) + 0x9E3779B9u;   // deterministic hash RNG
    x ^= x >> 16; x *= 0x85EBCA6Bu; x ^= x >> 13; x *= 0xC2B2AE35u; x ^= x >> 16;
    float u01 = (float)(x >> 8) * (1.0f / 16777216.0f);
    float gum = -logf(-logf(u01 + 1e-20f) + 1e-20f);
    float p = v + gum;
    if (p > best) { best = p; bi = a; }
  }
  float s = 0.0f;
  for (int a = 0; a < 16; ++a) s += expf(l[a] - m);
  logp_out[i] = l[bi] - m - logf(s);
  agent_out[i] = (float)bi;
}

// compact non-depot nodes in ascending order: pos(n) = n - #{depots < n}
__global__ __launch_bounds__(256) void k_compact(const float* logp, const int* depots,
                                                 float* out1, _Float16* clp16) {
  int b = blockIdx.x;
  int dep[16];
  for (int j = 0; j < 16; ++j) dep[j] = depots[b * NDd + j];
  for (int n = threadIdx.x; n < Nn; n += 256) {
    int cnt = 0; bool isdep = false;
    for (int j = 0; j < 16; ++j) { isdep |= (dep[j] == n); cnt += (dep[j] < n); }
    if (!isdep) {
      int pos = n - cnt;
      float v = logp[(long)b * Nn + n];
      out1[b * (Nn - NDd) + pos] = v;
      clp16[(long)b * Nn + pos] = (_Float16)v;    // cols 1008..1023 pre-zeroed
    }
  }
}

// surrogate head: out[b] = dot(s2[b,:], w) + bias
__global__ __launch_bounds__(256) void k_surfinal(const _Float16* s2, const float* w,
                                                  const float* bias, float* out) {
  __shared__ float red[256];
  int b = blockIdx.x, t = threadIdx.x;
  red[t] = (float)s2[(long)b * SHh + t] * w[t];
  __syncthreads();
  for (int s = 128; s > 0; s >>= 1) { if (t < s) red[t] += red[t + s]; __syncthreads(); }
  if (t == 0) out[b] = red[0] + bias[0];
}

// ---------------------------------------------------------------------------

extern "C" void kernel_launch(void* const* d_in, const int* in_sizes, int n_in,
                              void* d_out, int out_size, void* d_ws, size_t ws_size,
                              hipStream_t stream) {
  (void)in_sizes; (void)n_in; (void)out_size; (void)ws_size;
  // ----- inputs in setup_inputs() dict order (recursive insertion order) ---
  const float* coords = (const float*)d_in[0];
  const int*   depots = (const int*)d_in[1];
  const float* b1l1W = (const float*)d_in[2];  const float* b1l1b = (const float*)d_in[3];
  const float* b1g   = (const float*)d_in[4];  const float* b1b   = (const float*)d_in[5];
  const float* b1l2W = (const float*)d_in[6];  const float* b1l2b = (const float*)d_in[7];
  // gin[i]: l1.W, l1.b, l2.W, l2.b, bn.g, bn.b at 8+6i..13+6i
  const float* b2W  = (const float*)d_in[26];  const float* b2b  = (const float*)d_in[27];
  const float* b2g  = (const float*)d_in[28];  const float* b2bb = (const float*)d_in[29];
  const float* aeWkW = (const float*)d_in[30]; const float* aeWkb = (const float*)d_in[31];
  const float* aeWqW = (const float*)d_in[32]; const float* aeWqb = (const float*)d_in[33];
  const float* aeWvW = (const float*)d_in[34]; const float* aeWvb = (const float*)d_in[35];
  const float* aeWoW = (const float*)d_in[36]; const float* aeWob = (const float*)d_in[37];
  const float* alWkW = (const float*)d_in[38]; const float* alWkb = (const float*)d_in[39];
  const float* alWqW = (const float*)d_in[40]; const float* alWqb = (const float*)d_in[41];
  const float* s0W = (const float*)d_in[42];   const float* s0b = (const float*)d_in[43];
  const float* s1W = (const float*)d_in[44];   const float* s1b = (const float*)d_in[45];
  const float* s2W = (const float*)d_in[46];   const float* s2b = (const float*)d_in[47];

  float* out1 = (float*)d_out;                       // (B, 1008) customer log-probs
  float* out2 = out1 + (long)Bg * (Nn - NDd);        // (B,) predicted lengths
  float* out3 = out2 + Bg;                           // (B, N) agent_idx (as float)

  // ----- bump allocator on d_ws -----
  size_t off = 0;
  auto alloc = [&](size_t bytes) -> void* {
    void* p = (char*)d_ws + off;
    off = (off + bytes + 255) & ~(size_t)255;
    return p;
  };
  float*    bufA   = (float*)alloc(Mbig * 64 * 4);      // h
  float*    bufB   = (float*)alloc(Mbig * 64 * 4);      // v / ve
  float*    bufC   = (float*)alloc(Mbig * 64 * 4);      // pre-BN scratch
  _Float16* act16a = (_Float16*)alloc(Mbig * 64 * 2);   // f16 GEMM input / ve16
  _Float16* act16b = (_Float16*)alloc(Mbig * 64 * 2);
  _Float16* K16    = (_Float16*)alloc(Mbig * 64 * 2);
  _Float16* Vt16   = (_Float16*)alloc(Mbig * 64 * 2);   // V transposed (B, D, N)
  _Float16* K2_16  = (_Float16*)alloc(Mbig * 128 * 2);
  float*    Ubuf   = (float*)alloc((long)Bg * 16 * Nn * 4);   // U then logits
  _Float16* P16    = (_Float16*)alloc((long)Bg * 16 * Nn * 2);
  _Float16* fc16   = (_Float16*)alloc(4096L * 128 * 2);
  _Float16* Q16    = (_Float16*)alloc(4096L * 64 * 2);
  _Float16* Q2_16  = (_Float16*)alloc(4096L * 128 * 2);
  _Float16* apre16 = (_Float16*)alloc(4096L * 64 * 2);
  float*    agentF = (float*)alloc(4096L * 16 * 4);
  _Float16* agtP16 = (_Float16*)alloc(4096L * 32 * 2);
  float*    gsum   = (float*)alloc(Bg * 64 * 4);
  _Float16* ctx16  = (_Float16*)alloc(Bg * 64 * 2);
  float*    ctxlin = (float*)alloc(Bg * 64 * 4);
  float*    ctxbn  = (float*)alloc(Bg * 64 * 4);
  float*    stats  = (float*)alloc(256 * 4);            // sums|sumsq|mu|rstd
  float*    flag   = (float*)alloc(Mbig * 4);
  float*    logp   = (float*)alloc(Mbig * 4);
  _Float16* clp16  = (_Float16*)alloc((long)Bg * Nn * 2);  // padded to 1024
  _Float16* sA16   = (_Float16*)alloc((long)Bg * SHh * 2);
  _Float16* sB16   = (_Float16*)alloc((long)Bg * SHh * 2);
  // f16 packed weights (fragment order)
  _Float16* w_b1l2 = (_Float16*)alloc(64 * 64 * 2);
  _Float16* w_gl1[3]; _Float16* w_gl2[3];
  for (int i = 0; i < 3; ++i) { w_gl1[i] = (_Float16*)alloc(64 * 64 * 2);
                                w_gl2[i] = (_Float16*)alloc(64 * 64 * 2); }
  _Float16* w_b2   = (_Float16*)alloc(64 * 64 * 2);
  _Float16* w_aeWk = (_Float16*)alloc(64 * 64 * 2);
  _Float16* w_aeWq = (_Float16*)alloc(128 * 64 * 2);
  _Float16* w_aeWv = (_Float16*)alloc(64 * 64 * 2);
  _Float16* w_aeWo = (_Float16*)alloc(64 * 16 * 2);
  _Float16* w_alWk = (_Float16*)alloc(64 * 128 * 2);
  _Float16* w_alWqP= (_Float16*)alloc(32 * 128 * 2);     // K 16->32 padded
  _Float16* w_s0P  = (_Float16*)alloc(1024L * 256 * 2);  // K 1008->1024 padded
  _Float16* w_s1   = (_Float16*)alloc(256L * 256 * 2);

  auto blocks = [](long n, int bs) { return (unsigned)((n + bs - 1) / bs); };
  auto pack = [&](const float* s, _Float16* d, int Ksrc, int Kp, int Nc) {
    k_packW<<<blocks((long)Kp * Nc, 256), 256, 0, stream>>>(s, d, Ksrc, Kp, Nc);
  };
  auto stats_of = [&](const float* X, long M) {
    k_zero_f32<<<1, 128, 0, stream>>>(stats, 128);
    k_colstats<<<blocks(M, 1024), 64, 0, stream>>>(X, M, stats);
    k_bnfinal<<<1, 64, 0, stream>>>(stats, 1.0f / (float)M);
  };

  // ----- weight repack into WMMA fragment order -----
  pack(b1l2W, w_b1l2, 64, 64, 64);
  for (int i = 0; i < 3; ++i) {
    pack((const float*)d_in[8 + 6 * i],  w_gl1[i], 64, 64, 64);
    pack((const float*)d_in[10 + 6 * i], w_gl2[i], 64, 64, 64);
  }
  pack(b2W, w_b2, 64, 64, 64);
  pack(aeWkW, w_aeWk, 64, 64, 64);   pack(aeWqW, w_aeWq, 128, 128, 64);
  pack(aeWvW, w_aeWv, 64, 64, 64);   pack(aeWoW, w_aeWo, 64, 64, 16);
  pack(alWkW, w_alWk, 64, 64, 128);  pack(alWqW, w_alWqP, 16, 32, 128);
  pack(s0W, w_s0P, 1008, 1024, 256); pack(s1W, w_s1, 256, 256, 256);

  // ===== GraphEmbeddings block 1 =====
  k_zero_f32<<<blocks(Mbig, 256), 256, 0, stream>>>(flag, Mbig);
  k_flag<<<Bg, NDd, 0, stream>>>(depots, flag);
  k_lin1<<<blocks(Mbig * 64, 256), 256, 0, stream>>>(coords, flag, b1l1W, b1l1b,
                                                     bufC, Mbig * 64);
  stats_of(bufC, Mbig);
  k_bnapply<<<blocks(Mbig * 64, 256), 256, 0, stream>>>(bufC, stats, b1g, b1b,
      Mbig * 64, 1, nullptr, act16a, nullptr);                     // relu -> f16
  k_gemm<<<dim3(4096, 4), 128, 0, stream>>>(act16a, 64, w_b1l2, b1l2b,
                                            bufA, nullptr, 64, 64, 0, 0);   // h
  k_copy_f32<<<blocks(Mbig * 64, 256), 256, 0, stream>>>(bufA, bufB, Mbig * 64); // v=h

  // ===== 3x GIN layers =====
  for (int i = 0; i < 3; ++i) {
    const float* l1b_ = (const float*)d_in[9 + 6 * i];
    const float* l2b_ = (const float*)d_in[11 + 6 * i];
    const float* bng  = (const float*)d_in[12 + 6 * i];
    const float* bnb  = (const float*)d_in[13 + 6 * i];
    k_gsum<<<Bg, 64, 0, stream>>>(bufA, gsum);
    k_m16<<<blocks(Mbig * 64, 256), 256, 0, stream>>>(bufA, gsum, act16a, Mbig * 64);
    k_gemm<<<dim3(4096, 4), 128, 0, stream>>>(act16a, 64, w_gl1[i], l1b_,
                                              nullptr, act16b, 64, 64, 1, 0); // relu
    k_gemm<<<dim3(4096, 4), 128, 0, stream>>>(act16b, 64, w_gl2[i], l2b_,
                                              bufC, nullptr, 64, 64, 0, 0);
    stats_of(bufC, Mbig);
    k_bnapply<<<blocks(Mbig * 64, 256), 256, 0, stream>>>(bufC, stats, bng, bnb,
        Mbig * 64, 1, bufA, nullptr, bufB);         // h = relu(bn); v += h
  }

  // ===== context (global mean pool -> linear -> BN) =====
  k_cvt_f16<<<blocks(Mbig * 64, 256), 256, 0, stream>>>(bufB, act16a,
                                                        Mbig * 64, 1.0f);  // ve16
  k_gsum<<<Bg, 64, 0, stream>>>(bufB, gsum);
  k_cvt_f16<<<blocks(Bg * 64, 256), 256, 0, stream>>>(gsum, ctx16, Bg * 64,
                                                      1.0f / (float)Nn);
  k_gemm<<<dim3(4, 4), 128, 0, stream>>>(ctx16, 64, w_b2, b2b,
                                         ctxlin, nullptr, 64, 64, 0, 0);
  stats_of(ctxlin, Bg);
  k_bnapply<<<blocks(Bg * 64, 256), 256, 0, stream>>>(ctxlin, stats, b2g, b2bb,
      (long)Bg * 64, 0, ctxbn, nullptr, nullptr);

  // ===== agent embeddings (cross attention, depot queries) =====
  k_fc16<<<Bg * NDd, 128, 0, stream>>>(ctxbn, bufB, depots, fc16);
  k_gemm<<<dim3(64, 4), 128, 0, stream>>>(fc16, 128, w_aeWq, aeWqb,
                                          nullptr, Q16, 64, 128, 0, 0);
  k_gemm<<<dim3(4096, 4), 128, 0, stream>>>(act16a, 64, w_aeWk, aeWkb,
                                            nullptr, K16, 64, 64, 0, 0);
  k_gemm<<<dim3(4096, 4), 128, 0, stream>>>(act16a, 64, w_aeWv, aeWvb,
                                            nullptr, Vt16, 64, 64, 0, 1); // transposed
  k_attn_scores<<<dim3(Bg, 16), 128, 0, stream>>>(Q16, K16, Ubuf, 64, 0.125f, 0);
  k_softmax<<<Bg * 16, 256, 0, stream>>>(Ubuf, P16);
  k_attn_pv<<<Bg, 128, 0, stream>>>(P16, Vt16, apre16);
  k_gemm<<<dim3(64, 1), 128, 0, stream>>>(apre16, 64, w_aeWo, aeWob,
                                          agentF, nullptr, 16, 64, 0, 0);
  k_padcols<<<blocks(4096L * 32, 256), 256, 0, stream>>>(agentF, agtP16, 4096, 16, 32);

  // ===== allocation logits =====
  k_gemm<<<dim3(64, 8), 128, 0, stream>>>(agtP16, 32, w_alWqP, alWqb,
                                          nullptr, Q2_16, 128, 32, 0, 0);
  k_gemm<<<dim3(4096, 8), 128, 0, stream>>>(act16a, 64, w_alWk, alWkb,
                                            nullptr, K2_16, 128, 64, 0, 0);
  k_attn_scores<<<dim3(Bg, 16), 128, 0, stream>>>(Q2_16, K2_16, Ubuf, 128,
                                                  0.08838834764831845f, 1);
  k_mask<<<Bg, NDd, 0, stream>>>(Ubuf, depots);

  // ===== sampling, log-probs, customer compaction =====
  k_sample<<<blocks(Mbig, 256), 256, 0, stream>>>(Ubuf, out3, logp);
  k_zero_f16<<<blocks((long)Bg * Nn, 256), 256, 0, stream>>>(clp16, (long)Bg * Nn);
  k_compact<<<Bg, 256, 0, stream>>>(logp, depots, out1, clp16);

  // ===== surrogate network =====
  k_gemm<<<dim3(4, 16), 128, 0, stream>>>(clp16, 1024, w_s0P, s0b,
                                          nullptr, sA16, 256, 1024, 2, 0);  // tanh
  k_gemm<<<dim3(4, 16), 128, 0, stream>>>(sA16, 256, w_s1, s1b,
                                          nullptr, sB16, 256, 256, 2, 0);   // tanh
  k_surfinal<<<Bg, 256, 0, stream>>>(sB16, s2W, s2b, out2);
}
